// SemanticAlignmentModule_3307124818440
// MI455X (gfx1250) — compile-verified, matrix operand order checked by reading, not run on hardware
//
#include <hip/hip_runtime.h>
#include <math.h>

// Problem constants (from the reference):
#define ROWS   32768   // B*H*W
#define BATCH  8
#define NTOK   4096    // H*W
#define CV     256
#define CT     512
#define CH     1024    // 4*Cv
#define TTOK   100
#define TP     112     // text tokens padded to a multiple of 16 for WMMA tiles
#define TOPM   5

typedef __attribute__((ext_vector_type(16))) __bf16 v16bf;
typedef __attribute__((ext_vector_type(8)))  __bf16 v8bf;
typedef __attribute__((ext_vector_type(8)))  float  v8f;

// ---------- bf16 helpers (RNE), used for the hi/lo split ----------
__device__ __forceinline__ __bf16 f2bf(float f) {
  unsigned u = __builtin_bit_cast(unsigned, f);
  unsigned r = (u + 0x7fffu + ((u >> 16) & 1u)) >> 16;
  unsigned short s = (unsigned short)r;
  return __builtin_bit_cast(__bf16, s);
}
__device__ __forceinline__ float bf2f(__bf16 h) {
  unsigned short s = __builtin_bit_cast(unsigned short, h);
  unsigned u = ((unsigned)s) << 16;
  return __builtin_bit_cast(float, u);
}

// ---------- CDNA5 WMMA: D = A(16x32 bf16) * B(32x16 bf16) + C(f32) ----------
__device__ __forceinline__ v8f wmma_bf16(v16bf a, v16bf b, v8f c) {
  return __builtin_amdgcn_wmma_f32_16x16x32_bf16(
      /*neg_a=*/false, a, /*neg_b=*/false, b,
      /*c_mod=*/(short)0, c, /*reuse_a=*/false, /*reuse_b=*/false);
}

// Load one 16(M/N) x 32(K) bf16 fragment from a row-major tile (leading dim ld).
// ISA 7.12.2 16-bit A layout: lanes 0-15 -> row = lane, elems 0..7 = K 0..7,
// elems 8..15 = K 16..23; lanes 16-31 -> row = lane-16, K 8..15 / 24..31.
// B is stored transposed (N x K row-major) so the same loader serves both.
__device__ __forceinline__ v16bf load_frag(const __bf16* __restrict__ tile, int ld) {
  int lane = threadIdx.x & 31;
  const __bf16* p = tile + (size_t)(lane & 15) * ld + ((lane >> 4) << 3);
  v8bf a = *(const v8bf*)p;         // 16B aligned: ld and k0 are multiples of 32
  v8bf b = *(const v8bf*)(p + 16);
  return __builtin_shufflevector(a, b, 0,1,2,3,4,5,6,7,8,9,10,11,12,13,14,15);
}

__device__ __forceinline__ float block_sum(float v, float* red) {
  int tid = threadIdx.x;
  red[tid] = v; __syncthreads();
  for (int s = 128; s > 0; s >>= 1) {
    if (tid < s) red[tid] += red[tid + s];
    __syncthreads();
  }
  float r = red[0]; __syncthreads();
  return r;
}

// ---------- K0: weight transpose + bf16 hi/lo split: w(KxN) -> wT(NxK) ----------
__global__ void k_split_w(const float* __restrict__ w, int K, int N,
                          __bf16* __restrict__ thi, __bf16* __restrict__ tlo) {
  int i = blockIdx.x * blockDim.x + threadIdx.x;
  if (i >= K * N) return;
  int k = i / N, n = i % N;
  float v = w[i];
  __bf16 h = f2bf(v);
  thi[(size_t)n * K + k] = h;
  tlo[(size_t)n * K + k] = f2bf(v - bf2f(h));
}

// ---------- K1/K6: LayerNorm over Cv=256 (one row per 256-thread block). ----------
// Optionally fuses the sigmoid gate (K1 only).
__global__ void k_ln(const float* __restrict__ in,
                     const float* __restrict__ g, const float* __restrict__ b,
                     float* __restrict__ outf,
                     __bf16* __restrict__ ohi, __bf16* __restrict__ olo,
                     const float* __restrict__ gw, const float* __restrict__ gb,
                     float* __restrict__ gate) {
  __shared__ float red[256];
  int row = blockIdx.x, tid = threadIdx.x;
  float v = in[(size_t)row * CV + tid];
  float mean = block_sum(v, red) * (1.0f / CV);
  float d = v - mean;
  float var = block_sum(d * d, red) * (1.0f / CV);
  float x = d * rsqrtf(var + 1e-5f) * g[tid] + b[tid];
  outf[(size_t)row * CV + tid] = x;
  __bf16 h = f2bf(x);
  ohi[(size_t)row * CV + tid] = h;
  olo[(size_t)row * CV + tid] = f2bf(x - bf2f(h));
  if (gw) {
    float gp = block_sum(x * gw[tid], red);
    if (tid == 0) gate[row] = 1.0f / (1.0f + expf(-(gp + gb[0])));
  }
}

// ---------- K2: text prep: pad flags, k = l2norm(text) as bf16 hi/lo, v = text@vw+vb ----------
__global__ void k_text(const float* __restrict__ text,
                       const float* __restrict__ vw, const float* __restrict__ vb,
                       __bf16* __restrict__ khi, __bf16* __restrict__ klo,
                       float* __restrict__ vmat, int* __restrict__ padm) {
  __shared__ float red[256];
  int bt = blockIdx.x;                 // b*TP + t
  int b = bt / TP, t = bt % TP;
  int tid = threadIdx.x;
  const float* trow = text + ((size_t)b * TTOK + t) * CT;
  float e0 = 0.f, e1 = 0.f;
  if (t < TTOK) { e0 = trow[tid]; e1 = trow[tid + 256]; }
  float sabs = block_sum(fabsf(e0) + fabsf(e1), red);
  float ssq  = block_sum(e0 * e0 + e1 * e1, red);
  float rn = 1.0f / fmaxf(sqrtf(ssq), 1e-6f);
  float k0 = e0 * rn, k1 = e1 * rn;
  __bf16 h0 = f2bf(k0), h1 = f2bf(k1);
  size_t kb = (size_t)bt * CT;
  khi[kb + tid]       = h0;  klo[kb + tid]       = f2bf(k0 - bf2f(h0));
  khi[kb + tid + 256] = h1;  klo[kb + tid + 256] = f2bf(k1 - bf2f(h1));
  // v row (length Cv): trivial FLOPs, vw (512KB) stays hot in L2
  float acc = vb[tid];
  if (t < TTOK) {
    for (int k = 0; k < CT; ++k) acc += trow[k] * vw[(size_t)k * CV + tid];
  }
  vmat[(size_t)bt * CV + tid] = acc;
  if (tid == 0) padm[bt] = (t >= TTOK) || (sabs <= 1e-6f);
}

// ---------- K3: q = l2norm(x@qw + qb) * scale, output bf16 hi/lo ----------
// Block = 16 rows x 512 cols (8 waves x 4 tiles), epilogue via LDS.
__global__ void k_gemm_q(const __bf16* __restrict__ xhi, const __bf16* __restrict__ xlo,
                         const __bf16* __restrict__ wthi, const __bf16* __restrict__ wtlo,
                         const float* __restrict__ qb, const float* __restrict__ lsp,
                         __bf16* __restrict__ qhi, __bf16* __restrict__ qlo) {
  __shared__ float qt[16][CT + 4];
  __shared__ float part[16][16];
  __shared__ float rsc[16];
  int row0 = blockIdx.x * 16;
  int tid = threadIdx.x, wave = tid >> 5;
  int col0 = wave * 64;
  v8f acc[4] = {};
  for (int k0 = 0; k0 < CV; k0 += 32) {
    v16bf ah = load_frag(xhi + (size_t)row0 * CV + k0, CV);
    v16bf al = load_frag(xlo + (size_t)row0 * CV + k0, CV);
    for (int t = 0; t < 4; ++t) {
      v16bf bh = load_frag(wthi + (size_t)(col0 + t * 16) * CV + k0, CV);
      v16bf bl = load_frag(wtlo + (size_t)(col0 + t * 16) * CV + k0, CV);
      acc[t] = wmma_bf16(ah, bh, acc[t]);   // hi*hi
      acc[t] = wmma_bf16(ah, bl, acc[t]);   // hi*lo
      acc[t] = wmma_bf16(al, bh, acc[t]);   // lo*hi
    }
  }
  int lane = tid & 31, cc = lane & 15, rb = (lane >> 4) << 3;
  for (int t = 0; t < 4; ++t)
    for (int r = 0; r < 8; ++r) {
      int c = col0 + t * 16 + cc;
      qt[rb + r][c] = acc[t][r] + qb[c];
    }
  __syncthreads();
  int r = tid >> 4, cb = (tid & 15) * 32;
  float s = 0.f;
  for (int i = 0; i < 32; ++i) { float q = qt[r][cb + i]; s += q * q; }
  part[r][tid & 15] = s;
  __syncthreads();
  if (tid < 16) {
    float tot = 0.f;
    for (int i = 0; i < 16; ++i) tot += part[tid][i];
    float ls = fminf(fmaxf(lsp[0], -2.f), 2.f);
    float scale = expf(ls) * 0.044194173824159216f;   // 1/sqrt(Ct)
    rsc[tid] = scale / fmaxf(sqrtf(tot), 1e-6f);      // fold scale into q
  }
  __syncthreads();
  for (int i = 0; i < 32; ++i) {
    int c = cb + i;
    float q = qt[r][c] * rsc[r];
    __bf16 h = f2bf(q);
    qhi[(size_t)(row0 + r) * CT + c] = h;
    qlo[(size_t)(row0 + r) * CT + c] = f2bf(q - bf2f(h));
  }
}

// ---------- K4: sim = q@k^T (16 rows x 112 cols), fused top-5 / softmax / gather ----------
// y = x + alpha*gate*aligned written in place over x.
__global__ void k_sim_topk(const __bf16* __restrict__ qhi, const __bf16* __restrict__ qlo,
                           const __bf16* __restrict__ khi, const __bf16* __restrict__ klo,
                           const float* __restrict__ vmat, const int* __restrict__ padm,
                           const float* __restrict__ gate, const float* __restrict__ alphap,
                           float* __restrict__ xy) {
  __shared__ float sim[16][TP + 1];
  __shared__ int   padf[TP];
  __shared__ float attnS[16][TOPM];
  __shared__ int   idxS[16][TOPM];
  int row0 = blockIdx.x * 16;
  int bb = row0 / NTOK;                 // 16 | NTOK, so one batch per block
  int tid = threadIdx.x, wave = tid >> 5;
  if (tid < TP) padf[tid] = padm[bb * TP + tid];
  if (wave < 7) {                       // 7 column tiles of 16 cover TP=112
    v8f acc = {};
    int c0 = wave * 16;
    for (int k0 = 0; k0 < CT; k0 += 32) {
      v16bf ah = load_frag(qhi + (size_t)row0 * CT + k0, CT);
      v16bf al = load_frag(qlo + (size_t)row0 * CT + k0, CT);
      v16bf bh = load_frag(khi + (size_t)(bb * TP + c0) * CT + k0, CT);
      v16bf bl = load_frag(klo + (size_t)(bb * TP + c0) * CT + k0, CT);
      acc = wmma_bf16(ah, bh, acc);
      acc = wmma_bf16(ah, bl, acc);
      acc = wmma_bf16(al, bh, acc);
    }
    int lane = tid & 31, cc = lane & 15, rb = (lane >> 4) << 3;
    for (int r = 0; r < 8; ++r) sim[rb + r][c0 + cc] = acc[r];
  }
  __syncthreads();
  if (tid < 16) {
    // top-5 over t<100 only (fake tiles t>=100 excluded); ties -> lowest index
    float tv[TOPM]; int ti[TOPM];
    for (int m = 0; m < TOPM; ++m) { tv[m] = -__builtin_inff(); ti[m] = 0; }
    for (int t = 0; t < TTOK; ++t) {
      float vl = sim[tid][t];
      if (vl > tv[TOPM - 1]) {
        int p = TOPM - 1;
        while (p > 0 && vl > tv[p - 1]) { tv[p] = tv[p - 1]; ti[p] = ti[p - 1]; --p; }
        tv[p] = vl; ti[p] = t;
      }
    }
    for (int m = 0; m < TOPM; ++m)
      if (padf[ti[m]]) tv[m] = -__builtin_inff();     // mask pads AFTER selection
    float mx = tv[0];
    for (int m = 1; m < TOPM; ++m) mx = fmaxf(mx, tv[m]);
    float w[TOPM], sum = 0.f;
    for (int m = 0; m < TOPM; ++m) {
      w[m] = (tv[m] == -__builtin_inff()) ? 0.f : expf(tv[m] - mx);
      sum += w[m];
    }
    float inv = 1.0f / sum;
    for (int m = 0; m < TOPM; ++m) { attnS[tid][m] = w[m] * inv; idxS[tid][m] = ti[m]; }
  }
  __syncthreads();
  int r = tid >> 4, l = tid & 15;
  int gr = row0 + r;
  float ga = gate[gr] * alphap[0];
  float aw[TOPM]; int ai[TOPM];
  for (int m = 0; m < TOPM; ++m) { aw[m] = attnS[r][m]; ai[m] = idxS[r][m]; }
  for (int j = 0; j < CV / 16; ++j) {
    int c = j * 16 + l;
    float a = 0.f;
    for (int m = 0; m < TOPM; ++m)
      a += aw[m] * vmat[(size_t)(bb * TP + ai[m]) * CV + c];
    xy[(size_t)gr * CV + c] += ga * a;   // y = x + alpha*gate*aligned (in place)
  }
}

// ---------- K7: h = gelu(yn@w1 + b1) as bf16 hi/lo. Block = 32x256 (8 waves x 4 tiles) ----------
__global__ void k_mlp1(const __bf16* __restrict__ ahi, const __bf16* __restrict__ alo,
                       const __bf16* __restrict__ wthi, const __bf16* __restrict__ wtlo,
                       const float* __restrict__ b1,
                       __bf16* __restrict__ hhi, __bf16* __restrict__ hlo) {
  int tid = threadIdx.x, wave = tid >> 5;
  int row0 = blockIdx.x * 32 + (wave >> 2) * 16;
  int col0 = blockIdx.y * 256 + (wave & 3) * 64;
  v8f acc[4] = {};
  for (int k0 = 0; k0 < CV; k0 += 32) {
    v16bf ah = load_frag(ahi + (size_t)row0 * CV + k0, CV);
    v16bf al = load_frag(alo + (size_t)row0 * CV + k0, CV);
    for (int t = 0; t < 4; ++t) {
      v16bf bh = load_frag(wthi + (size_t)(col0 + t * 16) * CV + k0, CV);
      v16bf bl = load_frag(wtlo + (size_t)(col0 + t * 16) * CV + k0, CV);
      acc[t] = wmma_bf16(ah, bh, acc[t]);
      acc[t] = wmma_bf16(ah, bl, acc[t]);
      acc[t] = wmma_bf16(al, bh, acc[t]);
    }
  }
  int lane = tid & 31, cc = lane & 15, rb = (lane >> 4) << 3;
  for (int t = 0; t < 4; ++t)
    for (int r = 0; r < 8; ++r) {
      int row = row0 + rb + r, c = col0 + t * 16 + cc;
      float v = acc[t][r] + b1[c];
      float ge = 0.5f * v * (1.0f + erff(v * 0.70710678118654752f)); // exact gelu
      __bf16 h = f2bf(ge);
      hhi[(size_t)row * CH + c] = h;
      hlo[(size_t)row * CH + c] = f2bf(ge - bf2f(h));
    }
}

// ---------- K8: out = yn + h@w2 + b2. Block = 32x256 covers all Cv columns ----------
__global__ void k_mlp2(const __bf16* __restrict__ ahi, const __bf16* __restrict__ alo,
                       const __bf16* __restrict__ wthi, const __bf16* __restrict__ wtlo,
                       const float* __restrict__ b2, const float* __restrict__ yn,
                       float* __restrict__ out) {
  int tid = threadIdx.x, wave = tid >> 5;
  int row0 = blockIdx.x * 32 + (wave >> 2) * 16;
  int col0 = (wave & 3) * 64;
  v8f acc[4] = {};
  for (int k0 = 0; k0 < CH; k0 += 32) {
    v16bf ah = load_frag(ahi + (size_t)row0 * CH + k0, CH);
    v16bf al = load_frag(alo + (size_t)row0 * CH + k0, CH);
    for (int t = 0; t < 4; ++t) {
      v16bf bh = load_frag(wthi + (size_t)(col0 + t * 16) * CH + k0, CH);
      v16bf bl = load_frag(wtlo + (size_t)(col0 + t * 16) * CH + k0, CH);
      acc[t] = wmma_bf16(ah, bh, acc[t]);
      acc[t] = wmma_bf16(ah, bl, acc[t]);
      acc[t] = wmma_bf16(al, bh, acc[t]);
    }
  }
  int lane = tid & 31, cc = lane & 15, rb = (lane >> 4) << 3;
  for (int t = 0; t < 4; ++t)
    for (int r = 0; r < 8; ++r) {
      int row = row0 + rb + r, c = col0 + t * 16 + cc;
      out[(size_t)row * CV + c] = yn[(size_t)row * CV + c] + acc[t][r] + b2[c];
    }
}

extern "C" void kernel_launch(void* const* d_in, const int* in_sizes, int n_in,
                              void* d_out, int out_size, void* d_ws, size_t ws_size,
                              hipStream_t stream) {
  (void)in_sizes; (void)n_in; (void)out_size; (void)ws_size;
  const float* visual = (const float*)d_in[0];
  const float* text   = (const float*)d_in[1];
  const float* ln1_g  = (const float*)d_in[2];
  const float* ln1_b  = (const float*)d_in[3];
  const float* qw     = (const float*)d_in[4];
  const float* qb     = (const float*)d_in[5];
  const float* vw     = (const float*)d_in[6];
  const float* vb     = (const float*)d_in[7];
  const float* gate_w = (const float*)d_in[8];
  const float* gate_b = (const float*)d_in[9];
  const float* lsc    = (const float*)d_in[10];
  const float* alpha  = (const float*)d_in[11];
  const float* ln2_g  = (const float*)d_in[12];
  const float* ln2_b  = (const float*)d_in[13];
  const float* w1     = (const float*)d_in[14];
  const float* b1     = (const float*)d_in[15];
  const float* w2     = (const float*)d_in[16];
  const float* b2     = (const float*)d_in[17];
  // d_in[18] = top_m (hardcoded TOPM=5)

  // Bump allocator over d_ws (total ~207 MB with lifetime reuse)
  char* ws = (char*)d_ws;
  size_t off = 0;
  auto alloc = [&](size_t bytes) {
    size_t o = off;
    off = (off + bytes + 255) & ~(size_t)255;
    return o;
  };
  // Region 0: x (f32 + bf16 hi/lo) -- dead after K4/K6, reused as h_lo (exactly 64 MiB)
  size_t off_xf   = alloc((size_t)ROWS * CV * 4);
  size_t off_xhi  = alloc((size_t)ROWS * CV * 2);
  size_t off_xlo  = alloc((size_t)ROWS * CV * 2);
  // Region 1: q hi/lo -- dead after K4, reused as h_hi (exactly 64 MiB)
  size_t off_q    = alloc((size_t)ROWS * CT * 2 * 2);
  size_t off_gate = alloc((size_t)ROWS * 4);
  size_t off_khi  = alloc((size_t)BATCH * TP * CT * 2);
  size_t off_klo  = alloc((size_t)BATCH * TP * CT * 2);
  size_t off_v    = alloc((size_t)BATCH * TP * CV * 4);
  size_t off_pad  = alloc((size_t)BATCH * TP * 4);
  size_t off_yn   = alloc((size_t)ROWS * CV * 4);
  size_t off_ynhi = alloc((size_t)ROWS * CV * 2);
  size_t off_ynlo = alloc((size_t)ROWS * CV * 2);
  size_t off_qwt  = alloc((size_t)CV * CT * 2 * 2);
  size_t off_w1t  = alloc((size_t)CV * CH * 2 * 2);
  size_t off_w2t  = alloc((size_t)CH * CV * 2 * 2);

  float*  xf    = (float*)(ws + off_xf);
  __bf16* xhi   = (__bf16*)(ws + off_xhi);
  __bf16* xlo   = (__bf16*)(ws + off_xlo);
  __bf16* qhi   = (__bf16*)(ws + off_q);
  __bf16* qlo   = qhi + (size_t)ROWS * CT;
  float*  gate  = (float*)(ws + off_gate);
  __bf16* khi   = (__bf16*)(ws + off_khi);
  __bf16* klo   = (__bf16*)(ws + off_klo);
  float*  vmat  = (float*)(ws + off_v);
  int*    padm  = (int*)(ws + off_pad);
  float*  yn    = (float*)(ws + off_yn);
  __bf16* ynhi  = (__bf16*)(ws + off_ynhi);
  __bf16* ynlo  = (__bf16*)(ws + off_ynlo);
  __bf16* qwthi = (__bf16*)(ws + off_qwt);
  __bf16* qwtlo = qwthi + (size_t)CV * CT;
  __bf16* w1thi = (__bf16*)(ws + off_w1t);
  __bf16* w1tlo = w1thi + (size_t)CV * CH;
  __bf16* w2thi = (__bf16*)(ws + off_w2t);
  __bf16* w2tlo = w2thi + (size_t)CH * CV;
  // Lifetime reuse: h (hi/lo) overlays the q and x regions (both dead by K7)
  __bf16* hhi = (__bf16*)(ws + off_q);    // 64 MiB == ROWS*CH*2
  __bf16* hlo = (__bf16*)(ws + off_xf);   // 64 MiB == ROWS*CH*2

  // K0: weight transpose + bf16 split
  k_split_w<<<(CV * CT + 255) / 256, 256, 0, stream>>>(qw, CV, CT, qwthi, qwtlo);
  k_split_w<<<(CV * CH + 255) / 256, 256, 0, stream>>>(w1, CV, CH, w1thi, w1tlo);
  k_split_w<<<(CH * CV + 255) / 256, 256, 0, stream>>>(w2, CH, CV, w2thi, w2tlo);
  // K1: LN1 + gate
  k_ln<<<ROWS, 256, 0, stream>>>(visual, ln1_g, ln1_b, xf, xhi, xlo,
                                 gate_w, gate_b, gate);
  // K2: text prep (k, v, pad)
  k_text<<<BATCH * TP, 256, 0, stream>>>(text, vw, vb, khi, klo, vmat, padm);
  // K3: q GEMM + l2norm + scale
  k_gemm_q<<<ROWS / 16, 256, 0, stream>>>(xhi, xlo, qwthi, qwtlo, qb, lsc, qhi, qlo);
  // K4: sim GEMM + top-5 + softmax + gather + gated residual (y in place over x)
  k_sim_topk<<<ROWS / 16, 256, 0, stream>>>(qhi, qlo, khi, klo, vmat, padm,
                                            gate, alpha, xf);
  // K6: LN2
  k_ln<<<ROWS, 256, 0, stream>>>(xf, ln2_g, ln2_b, yn, ynhi, ynlo,
                                 nullptr, nullptr, nullptr);
  // K7: MLP up + GELU
  k_mlp1<<<dim3(ROWS / 32, CH / 256), 256, 0, stream>>>(ynhi, ynlo, w1thi, w1tlo,
                                                        b1, hhi, hlo);
  // K8: MLP down + residual -> out
  k_mlp2<<<ROWS / 32, 256, 0, stream>>>(hhi, hlo, w2thi, w2tlo, b2, yn,
                                        (float*)d_out);
}
